// GPTBlock_1958505087721
// MI455X (gfx1250) — matrix operand order repaired and checked by reading, not compiled, not run on hardware
//
#include <hip/hip_runtime.h>
#include <hip/hip_bf16.h>

typedef unsigned short u16;
typedef __attribute__((ext_vector_type(16))) __bf16 v16bf;
typedef __attribute__((ext_vector_type(8)))  __bf16 v8bf;
typedef __attribute__((ext_vector_type(8)))  float  v8f;
typedef __attribute__((ext_vector_type(4)))  unsigned int u32x4;
typedef __attribute__((ext_vector_type(8)))  int i32x8;
typedef __attribute__((ext_vector_type(4)))  int i32x4;

#define DEV __device__ __forceinline__

static constexpr int BATCH = 4, T = 2048, CDIM = 1024, NH = 16, DH = 64, FF = 4096;
static constexpr int MTOK = BATCH * T;   // 8192 rows

DEV u16 f2bf(float f) {
  __hip_bfloat16 h = __float2bfloat16(f);
  return __builtin_bit_cast(u16, h);
}

union Frag { v16bf v; v8bf h[2]; };

DEV float rmax16(float x) {
#pragma unroll
  for (int m = 1; m < 16; m <<= 1) x = fmaxf(x, __shfl_xor(x, m, 32));
  return x;
}
DEV float rsum16(float x) {
#pragma unroll
  for (int m = 1; m < 16; m <<= 1) x += __shfl_xor(x, m, 32);
  return x;
}

// LDS byte offset: flat shared address keeps the LDS offset in addr[31:0] (ISA 10.2 aperture map)
DEV unsigned lds_off(const void* p) { return (unsigned)(size_t)p; }

// ---- Tensor Data Mover: DMA one 128-row x 32-col bf16 tile (row pitch = kstride elems)
// into LDS with 16B padding after every 64B row -> 80B LDS pitch (LDT=40 halves).
// D# per CDNA5 ISA. Issued wave-level; tracked by TENSORcnt.
// This toolchain uses the 6-arg builtin: (u32x4, i32x8, i32x4, i32x4, i32x8, i32 cpol).
DEV void tdm_tile(const u16* gsrc, unsigned ldsoff, int kstride) {
  unsigned long long ga = (unsigned long long)(size_t)gsrc;
  u32x4 g0;
  g0[0] = 1u;                                   // count=1 (valid descriptor)
  g0[1] = ldsoff;                               // lds_addr
  g0[2] = (unsigned)ga;                         // global_addr[31:0]
  g0[3] = ((unsigned)(ga >> 32) & 0x01FFFFFFu)  // global_addr[56:32]
          | (2u << 30);                         // type=2 ("image")
  i32x8 g1;
  g1[0] = (1 << 16)       // data_size = 2 bytes
        | (1 << 20)       // pad_enable
        | (3 << 22)       // pad_interval: 16 DWORDs (=64B row)
        | (3 << 25);      // pad_amount: 4 DWORDs (=16B) -> 80B pitch
  g1[1] = (32 << 16);     // tensor_dim0 = 32 (bits 63:48)
  g1[2] = (128 << 16);    // tensor_dim0 hi = 0; tensor_dim1 = 128 (bits 95:80)
  g1[3] = (32 << 16);     // tensor_dim1 hi = 0; tile_dim0 = 32 (bits 127:112)
  g1[4] = 128;            // tile_dim1 = 128; tile_dim2 = 0
  g1[5] = kstride;        // tensor_dim0_stride[31:0] = row pitch (elements)
  g1[6] = 0;              // stride hi, tensor_dim1_stride lo
  g1[7] = 0;
  i32x4 z4 = (i32x4)0;    // groups 2/3 unused (2D tile)
  i32x8 z8 = (i32x8)0;    // extra group (clang-23 6-arg form)
  __builtin_amdgcn_tensor_load_to_lds(g0, g1, z4, z4, z8, 0);
}

// ---------------- elementwise f32 -> bf16 ----------------
__global__ void k_f2bf(const float* __restrict__ x, u16* __restrict__ y, int n) {
  int i = blockIdx.x * blockDim.x + threadIdx.x;
  if (i < n) y[i] = f2bf(x[i]);
}

// ------------- transpose+convert: w[K][N] f32 -> wt[N][K] bf16 -------------
__global__ __launch_bounds__(256) void k_transpose_bf16(const float* __restrict__ w,
                                                        u16* __restrict__ wt, int K, int N) {
  __shared__ float tile[32][33];
  int n0 = blockIdx.x * 32, k0 = blockIdx.y * 32;
  int tx = threadIdx.x & 31, ty = threadIdx.x >> 5;
#pragma unroll
  for (int r = ty; r < 32; r += 8)
    tile[r][tx] = w[(size_t)(k0 + r) * N + n0 + tx];
  __syncthreads();
#pragma unroll
  for (int r = ty; r < 32; r += 8)
    wt[(size_t)(n0 + r) * K + k0 + tx] = f2bf(tile[tx][r]);
}

// ---------------- split fused qkv (bf16) into per-head Q, K, V^T ----------------
__global__ void k_split_qkv(const u16* __restrict__ qkv, u16* __restrict__ q,
                            u16* __restrict__ k, u16* __restrict__ vt) {
  size_t idx = (size_t)blockIdx.x * blockDim.x + threadIdx.x;
  int c = (int)(idx % (3 * CDIM));
  size_t bt = idx / (3 * CDIM);
  int t = (int)(bt % T), b = (int)(bt / T);
  int which = c >> 10, cc = c & 1023, h = cc >> 6, d = cc & 63;
  size_t bh = (size_t)b * NH + h;
  u16 val = qkv[idx];
  if (which == 0)      q [(bh * T + t) * DH + d] = val;
  else if (which == 1) k [(bh * T + t) * DH + d] = val;
  else                 vt[(bh * DH + d) * T + t] = val;   // V transposed per head
}

// ---------------- bf16 WMMA GEMM: C[M,N] = A[M,K] * B^T (B stored [N][K]) ----------------
// Tiles arrive via TDM (tensor_load_to_lds) into double-buffered LDS; wave 0 drives the DMA,
// TENSORcnt + workgroup barrier publish the tile. 8 waves, each computes a 64x32 sub-tile.
// EPI: 0 = bf16 out; 1 = +resid -> f32+bf16; 2 = GELU -> bf16; 3 = +resid -> f32
template <int EPI>
__global__ __launch_bounds__(256) void k_gemm(const u16* __restrict__ A, const u16* __restrict__ B,
                                              int N, int K,
                                              const float* __restrict__ resid,
                                              float* __restrict__ outf, u16* __restrict__ outbf) {
  constexpr int LDT = 40;                 // 32 halves + 8-half pad (matches TDM pad config)
  __shared__ u16 As[2][128 * LDT];
  __shared__ u16 Bs[2][128 * LDT];
  const int tid = threadIdx.x;
  const int wave = tid >> 5, lane = tid & 31;
  const int wm = wave >> 2, wn = wave & 3;          // 2 x 4 wave grid
  const int ll = lane & 15, lh = lane >> 4;
  const int m0 = blockIdx.y * 128, n0 = blockIdx.x * 128;
  const int nk = K >> 5;
  v8f acc[4][2] = {};

  const u16* aTile = A + (size_t)m0 * K;
  const u16* bTile = B + (size_t)n0 * K;

  if (wave == 0) {                                  // prologue: DMA tile 0
    tdm_tile(aTile, lds_off(As[0]), K);
    tdm_tile(bTile, lds_off(Bs[0]), K);
  }

  for (int kt = 0; kt < nk; kt++) {
    const int cur = kt & 1;
    if (wave == 0) {
      if (kt + 1 < nk) {                            // DMA next tile into other buffer
        const int nxt = cur ^ 1;
        tdm_tile(aTile + (kt + 1) * 32, lds_off(As[nxt]), K);
        tdm_tile(bTile + (kt + 1) * 32, lds_off(Bs[nxt]), K);
        __builtin_amdgcn_s_wait_tensorcnt(2);       // current tile's 2 DMAs done
      } else {
        __builtin_amdgcn_s_wait_tensorcnt(0);
      }
    }
    __syncthreads();                                // publish current tile to all waves

    const u16* as = As[cur];
    const u16* bs = Bs[cur];
    Frag bf[2];
#pragma unroll
    for (int nt = 0; nt < 2; nt++) {                // B frag: K 0-7/16-23 (lo) | 8-15/24-31 (hi)
      const u16* bp = &bs[(wn * 32 + nt * 16 + ll) * LDT + lh * 8];
      bf[nt].h[0] = *(const v8bf*)(bp);
      bf[nt].h[1] = *(const v8bf*)(bp + 16);
    }
#pragma unroll
    for (int mt = 0; mt < 4; mt++) {                // A frag: K 0-7/16-23 (lo) | 8-15/24-31 (hi)
      const u16* ap = &as[(wm * 64 + mt * 16 + ll) * LDT];
      Frag af;
      af.h[0] = *(const v8bf*)(ap + lh * 8);
      af.h[1] = *(const v8bf*)(ap + 16 + lh * 8);
#pragma unroll
      for (int nt = 0; nt < 2; nt++)
        acc[mt][nt] = __builtin_amdgcn_wmma_f32_16x16x32_bf16(
            false, af.v, false, bf[nt].v, (short)0, acc[mt][nt], false, false);
    }
    __syncthreads();                                // done reading before next DMA overwrites
  }

  // epilogue: C layout -> row = i + 8*laneHi, col = lane&15
#pragma unroll
  for (int mt = 0; mt < 4; mt++)
#pragma unroll
    for (int nt = 0; nt < 2; nt++)
#pragma unroll
      for (int i = 0; i < 8; i++) {
        int row = m0 + wm * 64 + mt * 16 + lh * 8 + i;
        int col = n0 + wn * 32 + nt * 16 + ll;
        size_t o = (size_t)row * N + col;
        float c = acc[mt][nt][i];
        if constexpr (EPI == 0) {
          outbf[o] = f2bf(c);
        } else if constexpr (EPI == 1) {
          float y = c + resid[o];
          outf[o] = y;
          outbf[o] = f2bf(y);
        } else if constexpr (EPI == 2) {
          float g = 0.5f * c * (1.0f + erff(c * 0.70710678118f));   // exact-erf GELU
          outbf[o] = f2bf(g);
        } else {
          outf[o] = c + resid[o];
        }
      }
}

// ---------------- flash attention: per block = one (b,h), 64 query rows ----------------
__global__ __launch_bounds__(128) void k_flash(const u16* __restrict__ qb, const u16* __restrict__ kb,
                                               const u16* __restrict__ vt, u16* __restrict__ ob) {
  constexpr int LDP = 40;
  __shared__ u16 Pb[4][16 * LDP];
  const int wave = threadIdx.x >> 5, lane = threadIdx.x & 31;
  const int ll = lane & 15, lh = lane >> 4;
  const int bh = blockIdx.y;
  const int b = bh >> 4, h = bh & 15;
  const int qbase = blockIdx.x * 64 + wave * 16;

  // Q A-fragments, resident for the whole loop (Dh=64 -> two K=32 chunks)
  const u16* qrow = qb + ((size_t)bh * T + qbase + ll) * DH;
  Frag qa[2];
#pragma unroll
  for (int kk = 0; kk < 2; kk++) {
    qa[kk].h[0] = *(const v8bf*)(qrow + kk * 32 + lh * 8);
    qa[kk].h[1] = *(const v8bf*)(qrow + kk * 32 + 16 + lh * 8);
  }

  v8f o[4] = {};
  float mrow[8], lrow[8];
#pragma unroll
  for (int i = 0; i < 8; i++) { mrow[i] = -__builtin_inff(); lrow[i] = 0.f; }
  u16* pw = Pb[wave];

  const int nblk = blockIdx.x * 2 + 2;   // causal: keys up to end of this 64-query tile
  for (int jb = 0; jb < nblk; jb++) {
    const int j0 = jb * 32;

    if (jb + 1 < nblk) {                 // prefetch next K/V block (global_prefetch_b8)
      __builtin_prefetch(kb + ((size_t)bh * T + j0 + 32) * DH, 0, 1);
      __builtin_prefetch(vt + ((size_t)bh * DH + ll) * T + j0 + 32, 0, 1);
    }

    // S = Q * K^T  (two 16-key N-tiles, two K=32 chunks over Dh)
    v8f s[2] = {};
#pragma unroll
    for (int nt = 0; nt < 2; nt++) {
      const u16* krow = kb + ((size_t)bh * T + j0 + nt * 16 + ll) * DH;
#pragma unroll
      for (int kk = 0; kk < 2; kk++) {
        Frag kf;                         // B frag: Dh 0-7/16-23 (lo) | 8-15/24-31 (hi)
        kf.h[0] = *(const v8bf*)(krow + kk * 32 + lh * 8);
        kf.h[1] = *(const v8bf*)(krow + kk * 32 + 16 + lh * 8);
        s[nt] = __builtin_amdgcn_wmma_f32_16x16x32_bf16(
            false, qa[kk].v, false, kf.v, (short)0, s[nt], false, false);
      }
    }

    // scale, causal mask, online softmax update
    float alpha[8];
#pragma unroll
    for (int i = 0; i < 8; i++) {
      const int q = qbase + lh * 8 + i;
      float s0 = s[0][i] * 0.125f; if (j0 + ll      > q) s0 = -__builtin_inff();
      float s1 = s[1][i] * 0.125f; if (j0 + 16 + ll > q) s1 = -__builtin_inff();
      float bmax = rmax16(fmaxf(s0, s1));
      float mn = fmaxf(mrow[i], bmax);
      alpha[i] = __expf(mrow[i] - mn);
      mrow[i] = mn;
      float p0 = __expf(s0 - mn), p1 = __expf(s1 - mn);
      lrow[i] = lrow[i] * alpha[i] + rsum16(p0 + p1);
      s[0][i] = p0; s[1][i] = p1;
    }
#pragma unroll
    for (int dt = 0; dt < 4; dt++)
#pragma unroll
      for (int i = 0; i < 8; i++) o[dt][i] *= alpha[i];

    // P: C-layout -> A-layout via per-wave LDS tile (bf16)
#pragma unroll
    for (int nt = 0; nt < 2; nt++)
#pragma unroll
      for (int i = 0; i < 8; i++)
        pw[(lh * 8 + i) * LDP + nt * 16 + ll] = f2bf(s[nt][i]);
    __syncthreads();
    Frag pf;
    pf.h[0] = *(const v8bf*)(pw + ll * LDP + lh * 8);
    pf.h[1] = *(const v8bf*)(pw + ll * LDP + 16 + lh * 8);

    // O += P * V  (V^T rows are key-contiguous -> clean B fragments)
#pragma unroll
    for (int dt = 0; dt < 4; dt++) {
      const u16* vr = vt + ((size_t)bh * DH + dt * 16 + ll) * T + j0 + lh * 8;
      Frag vf;
      vf.h[0] = *(const v8bf*)(vr);
      vf.h[1] = *(const v8bf*)(vr + 16);
      o[dt] = __builtin_amdgcn_wmma_f32_16x16x32_bf16(
          false, pf.v, false, vf.v, (short)0, o[dt], false, false);
    }
    __syncthreads();
  }

  // normalize and store to [B,T,H*Dh] bf16 (input to out-proj GEMM)
#pragma unroll
  for (int i = 0; i < 8; i++) lrow[i] = 1.0f / lrow[i];
#pragma unroll
  for (int dt = 0; dt < 4; dt++)
#pragma unroll
    for (int i = 0; i < 8; i++) {
      int q = qbase + lh * 8 + i;
      ob[((size_t)b * T + q) * CDIM + h * DH + dt * 16 + ll] = f2bf(o[dt][i] * lrow[i]);
    }
}

extern "C" void kernel_launch(void* const* d_in, const int* in_sizes, int n_in,
                              void* d_out, int out_size, void* d_ws, size_t ws_size,
                              hipStream_t stream) {
  (void)in_sizes; (void)n_in; (void)out_size; (void)ws_size;
  const float* x     = (const float*)d_in[0];
  const float* w_qkv = (const float*)d_in[1];
  const float* w_out = (const float*)d_in[2];
  const float* w_ff1 = (const float*)d_in[3];
  const float* w_ff2 = (const float*)d_in[4];

  char* p = (char*)d_ws;
  auto carve = [&](size_t bytes) { char* r = p; p += (bytes + 255) & ~(size_t)255; return r; };
  u16*   xbf    = (u16*)carve((size_t)MTOK * CDIM * 2);
  u16*   wqkvT  = (u16*)carve((size_t)3 * CDIM * CDIM * 2);
  u16*   woutT  = (u16*)carve((size_t)CDIM * CDIM * 2);
  u16*   wff1T  = (u16*)carve((size_t)FF * CDIM * 2);
  u16*   wff2T  = (u16*)carve((size_t)CDIM * FF * 2);
  u16*   qkvbf  = (u16*)carve((size_t)MTOK * 3 * CDIM * 2);
  u16*   qbf    = (u16*)carve((size_t)MTOK * CDIM * 2);
  u16*   kbf    = (u16*)carve((size_t)MTOK * CDIM * 2);
  u16*   vtbf   = (u16*)carve((size_t)MTOK * CDIM * 2);
  u16*   attnbf = (u16*)carve((size_t)MTOK * CDIM * 2);
  float* y1f    = (float*)carve((size_t)MTOK * CDIM * 4);
  u16*   y1bf   = (u16*)carve((size_t)MTOK * CDIM * 2);
  u16*   hbf    = (u16*)carve((size_t)MTOK * FF * 2);

  // stage inputs as bf16; weights transposed to [N][K]
  k_f2bf<<<(MTOK * CDIM) / 256, 256, 0, stream>>>(x, xbf, MTOK * CDIM);
  k_transpose_bf16<<<dim3(3 * CDIM / 32, CDIM / 32), 256, 0, stream>>>(w_qkv, wqkvT, CDIM, 3 * CDIM);
  k_transpose_bf16<<<dim3(CDIM / 32, CDIM / 32), 256, 0, stream>>>(w_out, woutT, CDIM, CDIM);
  k_transpose_bf16<<<dim3(FF / 32, CDIM / 32), 256, 0, stream>>>(w_ff1, wff1T, CDIM, FF);
  k_transpose_bf16<<<dim3(CDIM / 32, FF / 32), 256, 0, stream>>>(w_ff2, wff2T, FF, CDIM);

  // qkv = x @ w_qkv  (bf16 out)
  k_gemm<0><<<dim3(3 * CDIM / 128, MTOK / 128), 256, 0, stream>>>(
      xbf, wqkvT, 3 * CDIM, CDIM, nullptr, nullptr, qkvbf);
  k_split_qkv<<<(int)((size_t)MTOK * 3 * CDIM / 256), 256, 0, stream>>>(qkvbf, qbf, kbf, vtbf);

  // causal flash attention
  k_flash<<<dim3(T / 64, BATCH * NH), 128, 0, stream>>>(qbf, kbf, vtbf, attnbf);

  // y1 = x + attn @ w_out   (f32 + bf16 out)
  k_gemm<1><<<dim3(CDIM / 128, MTOK / 128), 256, 0, stream>>>(
      attnbf, woutT, CDIM, CDIM, x, y1f, y1bf);
  // h = gelu(y1 @ w_ff1)    (bf16 out)
  k_gemm<2><<<dim3(FF / 128, MTOK / 128), 256, 0, stream>>>(
      y1bf, wff1T, FF, CDIM, nullptr, nullptr, hbf);
  // out = y1 + h @ w_ff2    (f32 out)
  k_gemm<3><<<dim3(CDIM / 128, MTOK / 128), 256, 0, stream>>>(
      hbf, wff2T, CDIM, FF, y1f, (float*)d_out, nullptr);
}
